// AttentionDecoder_5403068858784
// MI455X (gfx1250) — matrix operand order repaired and checked by reading, not compiled, not run on hardware
//
#include <hip/hip_runtime.h>
#include <hip/hip_bf16.h>
#include <stdint.h>

// ---------------- problem constants ----------------
#define BB   128
#define TT   512
#define EE   512
#define HH   512
#define EMB  256
#define VV   100
#define KX   (EMB + EE + HH)   // 1280: concat [emb | context | h]
#define G4H  (4 * HH)          // 2048
#define VPAD 112               // V padded to multiple of 16

// ---------------- WMMA types ----------------
typedef __attribute__((ext_vector_type(16))) __bf16         bf16x16;
typedef __attribute__((ext_vector_type(16))) unsigned short ushort16;
typedef __attribute__((ext_vector_type(8)))  unsigned short ushort8;
typedef __attribute__((ext_vector_type(8)))  float          v8f;

union BFu {
  bf16x16  bf;
  ushort16 us;
  struct { ushort8 lo, hi; } h;
};

__device__ __forceinline__ unsigned short f2bf_bits(float f) {
  unsigned u = __float_as_uint(f);
  return (unsigned short)((u + 0x7FFFu + ((u >> 16) & 1u)) >> 16);  // RNE
}
__device__ __forceinline__ unsigned pack2bf(float a, float b) {
  return (unsigned)f2bf_bits(a) | ((unsigned)f2bf_bits(b) << 16);
}

// A fragment from bf16 row-major [M][lda] at (m0,k0).
// ISA 16-bit A layout: lane row = lane&15; per-lane K runs [c,c+8) and [c+16,c+24), c = 8*(lane>=16).
__device__ __forceinline__ bf16x16 load_a_bf16(const unsigned short* __restrict__ A, int lda,
                                               int m0, int k0, int lane) {
  int row = m0 + (lane & 15);
  int c   = k0 + ((lane >> 4) << 3);
  const unsigned short* p = A + (size_t)row * lda + c;
  BFu r;
  r.h.lo = *(const ushort8*)p;
  r.h.hi = *(const ushort8*)(p + 16);
  return r.bf;
}

// B fragment from pre-transposed bf16 Bt[N][ldb] at (k0,n0): per-lane K runs [c,c+8) and [c+16,c+24), c = 8*(lane>=16).
__device__ __forceinline__ bf16x16 load_b_bf16(const unsigned short* __restrict__ Bt, int ldb,
                                               int n0, int k0, int lane) {
  int nn = n0 + (lane & 15);
  int kk = k0 + ((lane >> 4) << 3);
  const unsigned short* p = Bt + (size_t)nn * ldb + kk;
  BFu r;
  r.h.lo = *(const ushort8*)p;
  r.h.hi = *(const ushort8*)(p + 16);
  return r.bf;
}

__device__ __forceinline__ v8f wmma_bf16(bf16x16 a, bf16x16 b, v8f c) {
  return __builtin_amdgcn_wmma_f32_16x16x32_bf16(false, a, false, b, (short)0, c, false, false);
}

__device__ __forceinline__ float sigmoidf_(float x) { return 1.0f / (1.0f + __expf(-x)); }

// ---------------- prep kernels ----------------
// dst[n*lddst + k] = bf16(src[k*ldsrc + n])   (transpose + convert)
__global__ void k_tcvt(const float* __restrict__ src, int ldsrc,
                       unsigned short* __restrict__ dst, int lddst, int K, int N) {
  int idx = blockIdx.x * 256 + threadIdx.x;
  if (idx >= K * N) return;
  int n = idx / K, k = idx - n * K;
  dst[(size_t)n * lddst + k] = f2bf_bits(src[(size_t)k * ldsrc + n]);
}

// straight f32 -> bf16 copy
__global__ void k_cvt(const float* __restrict__ src, unsigned short* __restrict__ dst, int n) {
  int i = blockIdx.x * 256 + threadIdx.x;
  if (i < n) dst[i] = f2bf_bits(src[i]);
}

__global__ void k_zero_us(unsigned short* __restrict__ p, int n) {
  int i = blockIdx.x * 256 + threadIdx.x;
  if (i < n) p[i] = 0;
}

__global__ void k_addvec(const float* __restrict__ a, const float* __restrict__ b,
                         float* __restrict__ o, int n) {
  int i = blockIdx.x * 256 + threadIdx.x;
  if (i < n) o[i] = a[i] + b[i];
}

// x_bf[b][0:256] = bf16(emb[tok[b]])
__global__ void k_emb(const float* __restrict__ emb, const int* __restrict__ tok,
                      unsigned short* __restrict__ x) {
  int idx = blockIdx.x * 256 + threadIdx.x;   // B*EMB
  int b = idx >> 8, j = idx & 255;
  x[(size_t)b * KX + j] = f2bf_bits(emb[(size_t)tok[b] * EMB + j]);
}

// x_bf[b][768:1280] = bf16(h[b])
__global__ void k_xh(const float* __restrict__ h, unsigned short* __restrict__ x) {
  int idx = blockIdx.x * 256 + threadIdx.x;   // B*H
  int b = idx >> 9, j = idx & 511;
  x[(size_t)b * KX + EMB + EE + j] = f2bf_bits(h[idx]);
}

// ---------------- generic WMMA GEMM: C[M,N] = A[M,K](bf16) x Bt[N,K](bf16) + bias ----------------
__global__ __launch_bounds__(256) void k_gemm(const unsigned short* __restrict__ A, int lda,
                                              const unsigned short* __restrict__ Bt, int ldb,
                                              const float* __restrict__ bias,
                                              float* __restrict__ C, int ldc,
                                              int K, int tiles_n, int total_tiles, int Nreal) {
  int lane = threadIdx.x & 31;
  int gid  = blockIdx.x * 8 + (threadIdx.x >> 5);
  if (gid >= total_tiles) return;
  int m0 = (gid / tiles_n) << 4;
  int n0 = (gid % tiles_n) << 4;
  v8f acc = {};
#pragma unroll 2
  for (int k = 0; k < K; k += 32) {
    bf16x16 a = load_a_bf16(A, lda, m0, k, lane);
    bf16x16 b = load_b_bf16(Bt, ldb, n0, k, lane);
    acc = wmma_bf16(a, b, acc);
  }
  int n = n0 + (lane & 15);
  if (n < Nreal) {
    float bn = bias ? bias[n] : 0.0f;
#pragma unroll
    for (int i = 0; i < 8; ++i) {
      int r = m0 + i + ((lane >> 4) << 3);
      C[(size_t)r * ldc + n] = acc[i] + bn;
    }
  }
}

// ---------------- fused attention scores ----------------
// scores[b,t] = sum_h tanh( (enc[b,t,:]@W_enc)[h] + pd[b,h] ) * v_w[h]
// One workgroup per 16-row (b,t)-tile; 8 waves x 4 N-tiles cover H=512.
// A tile (16x32 f32) is staged through LDS as bf16, converted ONCE per workgroup,
// double-buffered so staging of step k+1 overlaps WMMA of step k.
#define APITCH 56   // shorts per LDS row: 112B = 28 banks -> 16B-aligned, conflict-free b128 reads

__device__ __forceinline__ void stage_a(const float* __restrict__ enc, int m0, int k,
                                        unsigned short* __restrict__ buf, int tid) {
  int row = tid >> 4;               // 16 rows, 16 threads each
  int col = (tid & 15) << 1;        // 2 consecutive f32 per thread
  const float* p = enc + (size_t)(m0 + row) * EE + k + col;
  float2 v = *(const float2*)p;     // coalesced 8B per lane
  *(unsigned*)&buf[row * APITCH + col] = pack2bf(v.x, v.y);
}

__global__ __launch_bounds__(256) void k_scores(const float* __restrict__ enc,
                                                const unsigned short* __restrict__ WtEnc,
                                                const float* __restrict__ pd,
                                                const float* __restrict__ vw,
                                                float* __restrict__ sc) {
  __shared__ unsigned short atile[2][16 * APITCH];
  __shared__ float srow[16];
  int tid = threadIdx.x, lane = tid & 31, w = tid >> 5;
  int m0 = blockIdx.x << 4;          // row in flattened [B*T]
  int b  = m0 >> 9;
  int t0 = m0 & 511;

  if (tid < 16) srow[tid] = 0.0f;
  stage_a(enc, m0, 0, atile[0], tid);
  __syncthreads();

  v8f acc[4];
#pragma unroll
  for (int j = 0; j < 4; ++j) acc[j] = (v8f){};

  int p = 0;
  for (int k = 0; k < EE; k += 32) {
    if (k + 32 < EE) stage_a(enc, m0, k + 32, atile[p ^ 1], tid);
    // A fragment from LDS (shared by all 8 waves)
    int r = lane & 15, c = (lane >> 4) << 3;
    const unsigned short* ap = &atile[p][r * APITCH + c];
    BFu af;
    af.h.lo = *(const ushort8*)ap;          // ds_read_b128
    af.h.hi = *(const ushort8*)(ap + 16);   // ds_read_b128
#pragma unroll
    for (int j = 0; j < 4; ++j) {
      int n0 = ((w << 2) + j) << 4;
      bf16x16 bb = load_b_bf16(WtEnc, EE, n0, k, lane);
      acc[j] = wmma_bf16(af.bf, bb, acc[j]);
    }
    __syncthreads();
    p ^= 1;
  }

  // epilogue: tanh + dot with v_w, reduce over the 16 columns of each tile
  float s[8];
#pragma unroll
  for (int i = 0; i < 8; ++i) s[i] = 0.0f;
#pragma unroll
  for (int j = 0; j < 4; ++j) {
    int n = (((w << 2) + j) << 4) + (lane & 15);
    float pdn = pd[(size_t)b * HH + n];
    float vn  = vw[n];
#pragma unroll
    for (int i = 0; i < 8; ++i) s[i] += tanhf(acc[j][i] + pdn) * vn;
  }
  // butterfly within each 16-lane half (rows replicated across those lanes)
#pragma unroll
  for (int off = 8; off; off >>= 1) {
#pragma unroll
    for (int i = 0; i < 8; ++i) s[i] += __shfl_xor(s[i], off, 32);
  }
  if ((lane & 15) == 0) {
    int rbase = (lane >> 4) << 3;
#pragma unroll
    for (int i = 0; i < 8; ++i) atomicAdd(&srow[rbase + i], s[i]);
  }
  __syncthreads();
  if (tid < 16) sc[(size_t)b * TT + t0 + tid] = srow[tid];
}

// ---------------- softmax over T (in place) ----------------
__global__ void k_softmax(float* __restrict__ sc) {
  __shared__ float red[256];
  int b = blockIdx.x, tid = threadIdx.x;
  float v0 = sc[(size_t)b * TT + tid];
  float v1 = sc[(size_t)b * TT + 256 + tid];
  red[tid] = fmaxf(v0, v1);
  __syncthreads();
  for (int s = 128; s > 0; s >>= 1) { if (tid < s) red[tid] = fmaxf(red[tid], red[tid + s]); __syncthreads(); }
  float m = red[0];
  __syncthreads();
  float e0 = __expf(v0 - m), e1 = __expf(v1 - m);
  red[tid] = e0 + e1;
  __syncthreads();
  for (int s = 128; s > 0; s >>= 1) { if (tid < s) red[tid] += red[tid + s]; __syncthreads(); }
  float inv = 1.0f / red[0];
  sc[(size_t)b * TT + tid]       = e0 * inv;
  sc[(size_t)b * TT + 256 + tid] = e1 * inv;
}

// ---------------- context: x_bf[b][256:768] = bf16( sum_t attn[b,t] * enc[b,t,:] ) ----------------
__global__ void k_context(const float* __restrict__ enc, const float* __restrict__ attn,
                          unsigned short* __restrict__ x) {
  int b = blockIdx.x, e = threadIdx.x;   // 512 threads
  const float* ep = enc + ((size_t)b * TT) * EE + e;
  const float* ap = attn + (size_t)b * TT;
  float acc = 0.0f;
#pragma unroll 4
  for (int t = 0; t < TT; ++t) acc += ap[t] * ep[(size_t)t * EE];
  x[(size_t)b * KX + EMB + e] = f2bf_bits(acc);
}

// ---------------- LSTM elementwise ----------------
__global__ void k_lstm(const float* __restrict__ g, const float* __restrict__ c,
                       float* __restrict__ hN, float* __restrict__ cN,
                       unsigned short* __restrict__ hN_bf) {
  int idx = blockIdx.x * 256 + threadIdx.x;   // B*H
  int b = idx >> 9, j = idx & 511;
  const float* gb = g + (size_t)b * G4H;
  float i_ = sigmoidf_(gb[j]);
  float f_ = sigmoidf_(gb[HH + j]);
  float gg = tanhf(gb[2 * HH + j]);
  float o_ = sigmoidf_(gb[3 * HH + j]);
  float cn = f_ * c[idx] + i_ * gg;
  float hn = o_ * tanhf(cn);
  cN[idx] = cn;
  hN[idx] = hn;
  hN_bf[idx] = f2bf_bits(hn);
}

// ---------------- workspace layout (bytes, all 16B-aligned) ----------------
#define OFF_WTENC 0u                                   // 512*512*2
#define OFF_WTDEC (OFF_WTENC + 512u*512u*2u)           // 512*512*2
#define OFF_WTG   (OFF_WTDEC + 512u*512u*2u)           // 2048*1280*2
#define OFF_WTOUT (OFF_WTG + 2048u*1280u*2u)           // 112*512*2
#define OFF_HBF   (OFF_WTOUT + 112u*512u*2u)           // 128*512*2
#define OFF_XBF   (OFF_HBF + 128u*512u*2u)             // 128*1280*2
#define OFF_HNBF  (OFF_XBF + 128u*1280u*2u)            // 128*512*2
#define OFF_PD    (OFF_HNBF + 128u*512u*2u)            // 128*512*4
#define OFF_SC    (OFF_PD + 128u*512u*4u)              // 128*512*4
#define OFF_GATES (OFF_SC + 128u*512u*4u)              // 128*2048*4
#define OFF_BG    (OFF_GATES + 128u*2048u*4u)          // 2048*4
#define OFF_BPD   (OFF_BG + 2048u*4u)                  // 512*4

extern "C" void kernel_launch(void* const* d_in, const int* in_sizes, int n_in,
                              void* d_out, int out_size, void* d_ws, size_t ws_size,
                              hipStream_t stream) {
  (void)in_sizes; (void)n_in; (void)out_size; (void)ws_size;
  const float* enc   = (const float*)d_in[0];
  const int*   tok   = (const int*)  d_in[1];
  const float* h     = (const float*)d_in[2];
  const float* c     = (const float*)d_in[3];
  const float* emb   = (const float*)d_in[4];
  const float* W_enc = (const float*)d_in[5];
  const float* b_enc = (const float*)d_in[6];
  const float* W_dec = (const float*)d_in[7];
  const float* b_dec = (const float*)d_in[8];
  const float* v_w   = (const float*)d_in[9];
  const float* W_ih  = (const float*)d_in[10];
  const float* W_hh  = (const float*)d_in[11];
  const float* b_ih  = (const float*)d_in[12];
  const float* b_hh  = (const float*)d_in[13];
  const float* W_out = (const float*)d_in[14];
  const float* b_out = (const float*)d_in[15];

  char* ws = (char*)d_ws;
  unsigned short* WtEnc  = (unsigned short*)(ws + OFF_WTENC);
  unsigned short* WtDec  = (unsigned short*)(ws + OFF_WTDEC);
  unsigned short* WtG    = (unsigned short*)(ws + OFF_WTG);
  unsigned short* WtOut  = (unsigned short*)(ws + OFF_WTOUT);
  unsigned short* h_bf   = (unsigned short*)(ws + OFF_HBF);
  unsigned short* x_bf   = (unsigned short*)(ws + OFF_XBF);
  unsigned short* hn_bf  = (unsigned short*)(ws + OFF_HNBF);
  float* pd    = (float*)(ws + OFF_PD);
  float* sc    = (float*)(ws + OFF_SC);
  float* gates = (float*)(ws + OFF_GATES);
  float* bg    = (float*)(ws + OFF_BG);
  float* bpd   = (float*)(ws + OFF_BPD);

  float* logits = (float*)d_out;                   // [128,100]
  float* h_new  = logits + BB * VV;                // [128,512]
  float* c_new  = h_new + BB * HH;                 // [128,512]

  // ---- weight / operand prep (transpose + f32->bf16) ----
  k_tcvt<<<(512*512 + 255)/256, 256, 0, stream>>>(W_enc, HH, WtEnc, EE, EE, HH);
  k_tcvt<<<(512*512 + 255)/256, 256, 0, stream>>>(W_dec, HH, WtDec, HH, HH, HH);
  k_tcvt<<<(2048*768 + 255)/256, 256, 0, stream>>>(W_ih, G4H, WtG,            KX, EMB + EE, G4H);
  k_tcvt<<<(2048*512 + 255)/256, 256, 0, stream>>>(W_hh, G4H, WtG + (EMB + EE), KX, HH, G4H);
  k_zero_us<<<(VPAD*512 + 255)/256, 256, 0, stream>>>(WtOut, VPAD * HH);
  k_tcvt<<<(100*512 + 255)/256, 256, 0, stream>>>(W_out, VV, WtOut, HH, HH, VV);
  k_cvt<<<(BB*HH + 255)/256, 256, 0, stream>>>(h, h_bf, BB * HH);
  k_addvec<<<8, 256, 0, stream>>>(b_ih, b_hh, bg, G4H);
  k_addvec<<<2, 256, 0, stream>>>(b_dec, b_enc, bpd, HH);

  // ---- proj_dec = h @ W_dec + (b_dec + b_enc)  [128,512] ----
  k_gemm<<<32, 256, 0, stream>>>(h_bf, HH, WtDec, HH, bpd, pd, HH,
                                 HH, /*tiles_n=*/32, /*total=*/256, /*Nreal=*/HH);

  // ---- fused scores: tanh(enc@W_enc + pd) . v_w  [128,512] ----
  k_scores<<<(BB * TT) / 16, 256, 0, stream>>>(enc, WtEnc, pd, v_w, sc);

  // ---- softmax over T ----
  k_softmax<<<BB, 256, 0, stream>>>(sc);

  // ---- assemble x_bf = [emb_tok | context | h]  (bf16 [128,1280]) ----
  k_emb<<<(BB * EMB) / 256, 256, 0, stream>>>(emb, tok, x_bf);
  k_context<<<BB, 512, 0, stream>>>(enc, sc, x_bf);
  k_xh<<<(BB * HH) / 256, 256, 0, stream>>>(h, x_bf);

  // ---- gates = x @ [W_ih;W_hh] + b_ih + b_hh  [128,2048], K=1280 ----
  k_gemm<<<128, 256, 0, stream>>>(x_bf, KX, WtG, KX, bg, gates, G4H,
                                  KX, /*tiles_n=*/128, /*total=*/1024, /*Nreal=*/G4H);

  // ---- LSTM elementwise -> h_new, c_new (+ bf16 copy of h_new) ----
  k_lstm<<<(BB * HH) / 256, 256, 0, stream>>>(gates, c, h_new, c_new, hn_bf);

  // ---- logits = h_new @ W_out + b_out  [128,100] ----
  k_gemm<<<7, 256, 0, stream>>>(hn_bf, HH, WtOut, HH, b_out, logits, VV,
                                HH, /*tiles_n=*/7, /*total=*/56, /*Nreal=*/VV);
}
